// Block_47261820125190
// MI455X (gfx1250) — matrix operand order repaired and checked by reading, not compiled, not run on hardware
//
#include <hip/hip_runtime.h>
#include <stdint.h>

typedef __attribute__((ext_vector_type(16))) __bf16 v16bf;
typedef __attribute__((ext_vector_type(8)))  float  v8f;
typedef unsigned short u16;
typedef unsigned int   u32;

#define T_SEQ 16
#define C_DIM 256
#define H_NUM 4
#define HS_DIM 64
#define FF_DIM 1024

// cheap bf16: round-half-up, 1 VALU + d16_hi store
static __device__ __forceinline__ u16 f2bf(float f) {
  return (u16)((__float_as_uint(f) + 0x8000u) >> 16);
}

// two floats -> packed bf16 pair (lo = a), single v_perm_b32 after the adds
static __device__ __forceinline__ u32 pack2bf(float a, float b) {
  u32 ua = __float_as_uint(a) + 0x8000u;
  u32 ub = __float_as_uint(b) + 0x8000u;
  return __builtin_amdgcn_perm(ub, ua, 0x07060302u);
}

static __device__ __forceinline__ v8f wmma_bf16(v16bf a, v16bf b, v8f c) {
  // (neg_a, A, neg_b, B, c_mod, C, reuse_a, reuse_b)
  return __builtin_amdgcn_wmma_f32_16x16x32_bf16(false, a, false, b, (short)0, c,
                                                 false, false);
}

// A fragment (16x32 bf16): lane m=lane&15; K = kbase + 16h + (0..15)
static __device__ __forceinline__ v16bf frag_A(const u16* row, int kbase, int hf) {
  union { uint4 u[2]; v16bf v; } t;
  t.u[0] = *(const uint4*)(row + kbase + 16 * hf);
  t.u[1] = *(const uint4*)(row + kbase + 8 + 16 * hf);
  return t.v;
}

// B fragment (32x16 bf16) from [N][Ktot]: lane n fixed; K = kbase + 16h + (0..15)
static __device__ __forceinline__ v16bf frag_B(const u16* w, int n, int Ktot,
                                               int kbase, int hf) {
  const u16* p = w + (size_t)n * Ktot + kbase + 16 * hf;
  union { uint4 u[2]; v16bf v; } t;
  t.u[0] = *(const uint4*)(p);
  t.u[1] = *(const uint4*)(p + 8);
  return t.v;
}

// Wave-private LDS: same-wave DS ops are in-order; only stop compiler reordering.
#define LDS_FENCE() asm volatile("" ::: "memory")

// fp32 [mat][K][N] -> bf16 [mat][N][K]
__global__ void conv_transpose_bf16(const float* __restrict__ src, u16* __restrict__ dst,
                                    int K, int N, int total) {
  int t = blockIdx.x * blockDim.x + threadIdx.x;
  if (t >= total) return;
  int mn = K * N;
  int mat = t / mn;
  int rem = t - mat * mn;
  int k = rem / N;
  int n = rem - k * N;
  dst[(size_t)mat * mn + (size_t)n * K + k] = f2bf(src[t]);
}

__global__ __launch_bounds__(128) void block_fused(
    const float* __restrict__ x,
    const u16* __restrict__ qT, const u16* __restrict__ kT, const u16* __restrict__ vTw,
    const u16* __restrict__ woT, const u16* __restrict__ w1T, const u16* __restrict__ w2T,
    const float* __restrict__ bo, const float* __restrict__ b1, const float* __restrict__ b2,
    const float* __restrict__ g1, const float* __restrict__ be1,
    const float* __restrict__ g2, const float* __restrict__ be2,
    float* __restrict__ out) {
  __shared__ __align__(16) float xs[4][16][264];   // x / residual (f32)
  __shared__ __align__(16) u16   hb[4][16][264];   // LN output (A operand)
  __shared__ __align__(16) u16   ab[4][16][264];   // attn concat (A operand)
  __shared__ __align__(16) u16   qb[4][16][72];    // q rows [t][d]
  __shared__ __align__(16) u16   kb[4][16][72];    // k rows [s][d] (B for scores)
  __shared__ __align__(16) u16   vt[4][64][32];    // v^T [d][s pad32] (B for attn)
  __shared__ __align__(16) u16   pb[4][16][32];    // softmax probs [t][s pad32]
  __shared__ __align__(16) u16   actb[4][16][32];  // relu chunk [t][32]
  __shared__ __align__(16) float sc[4][16][17];    // scores staging

  const int wv   = threadIdx.x >> 5;
  const int lane = threadIdx.x & 31;
  const int hf   = lane >> 4;
  const int l16  = lane & 15;
  const int b    = blockIdx.x * 4 + wv;

  // zero the K-padding (s=16..31) once; only s=0..15 is rewritten per head
  for (int i = lane; i < 16 * 16; i += 32) pb[wv][i >> 4][16 + (i & 15)] = 0;
  for (int i = lane; i < 64 * 16; i += 32) vt[wv][i >> 4][16 + (i & 15)] = 0;

  // load x tile [16][256]
  {
    const float* xb = x + (size_t)b * (T_SEQ * C_DIM);
    for (int i = lane * 4; i < T_SEQ * C_DIM; i += 32 * 4) {
      float4 v = *(const float4*)(xb + i);
      *(float4*)&xs[wv][i >> 8][i & 255] = v;
    }
  }
  LDS_FENCE();

  // per-row LayerNorm: lanes 0..15 own rows (upper half duplicates compute)
  auto layernorm = [&](const float* g, const float* be) {
    const float* row = &xs[wv][l16][0];
    float s = 0.f, ss = 0.f;
    for (int c = 0; c < C_DIM; c += 4) {
      float4 v = *(const float4*)(row + c);
      s  += v.x + v.y + v.z + v.w;
      ss += v.x * v.x + v.y * v.y + v.z * v.z + v.w * v.w;
    }
    float mu   = s * (1.f / C_DIM);
    float var  = ss * (1.f / C_DIM) - mu * mu;
    float rstd = rsqrtf(var + 1e-5f);
    if (lane < 16) {
      for (int c = 0; c < C_DIM; c += 2) {
        float v0 = (row[c]     - mu) * rstd * g[c]     + be[c];
        float v1 = (row[c + 1] - mu) * rstd * g[c + 1] + be[c + 1];
        *(u32*)&hb[wv][l16][c] = pack2bf(v0, v1);
      }
    }
  };

  layernorm(g1, be1);
  LDS_FENCE();

  v16bf afr[8];  // hoisted K=256 A-row fragments (64 VGPRs), reused per phase

  // ---------------- attention ----------------
#pragma unroll
  for (int kk = 0; kk < 8; ++kk) afr[kk] = frag_A(&hb[wv][l16][0], kk * 32, hf);

  const float scl = 0.0625f;  // C^-0.5 (NOT head_size^-0.5, per reference)
  for (int h = 0; h < H_NUM; ++h) {
    const u16* Wqh = qT  + h * HS_DIM * C_DIM;
    const u16* Wkh = kT  + h * HS_DIM * C_DIM;
    const u16* Wvh = vTw + h * HS_DIM * C_DIM;

    // q,k,v = h @ W  ([16,256]x[256,64]); A fragments resident in registers
#pragma unroll
    for (int nt = 0; nt < 4; ++nt) {
      v8f qa = {}, ka = {}, va = {};
#pragma unroll
      for (int kk = 0; kk < 8; ++kk) {
        qa = wmma_bf16(afr[kk], frag_B(Wqh, nt * 16 + l16, 256, kk * 32, hf), qa);
        ka = wmma_bf16(afr[kk], frag_B(Wkh, nt * 16 + l16, 256, kk * 32, hf), ka);
        va = wmma_bf16(afr[kk], frag_B(Wvh, nt * 16 + l16, 256, kk * 32, hf), va);
      }
#pragma unroll
      for (int r = 0; r < 8; ++r) {
        int m = r + 8 * hf, d = nt * 16 + l16;
        qb[wv][m][d] = f2bf(qa[r]);
        kb[wv][m][d] = f2bf(ka[r]);
        vt[wv][d][m] = f2bf(va[r]);   // transposed for B operand of p@v
      }
    }
    LDS_FENCE();

    // scores = q @ k^T  (K=64)
    v8f sa = {};
    sa = wmma_bf16(frag_A(&qb[wv][l16][0], 0, hf),
                   frag_B(&kb[wv][0][0], l16, 72, 0, hf), sa);
    sa = wmma_bf16(frag_A(&qb[wv][l16][0], 32, hf),
                   frag_B(&kb[wv][0][0], l16, 72, 32, hf), sa);
#pragma unroll
    for (int r = 0; r < 8; ++r) sc[wv][r + 8 * hf][l16] = sa[r] * scl;
    LDS_FENCE();

    // causal softmax, one row per lane
    {
      const int row = l16;
      float e[16];
      float mx = -1e30f;
#pragma unroll
      for (int s2 = 0; s2 < 16; ++s2) {
        float v = (s2 <= row) ? sc[wv][row][s2] : -1e30f;
        e[s2] = v;
        mx = fmaxf(mx, v);
      }
      float sum = 0.f;
#pragma unroll
      for (int s2 = 0; s2 < 16; ++s2) {
        float p = (s2 <= row) ? __expf(e[s2] - mx) : 0.f;
        e[s2] = p;
        sum += p;
      }
      float inv = 1.f / sum;
      if (lane < 16) {
#pragma unroll
        for (int s2 = 0; s2 < 16; s2 += 2)
          *(u32*)&pb[wv][row][s2] = pack2bf(e[s2] * inv, e[s2 + 1] * inv);
      }
    }
    LDS_FENCE();

    // attn = p @ v  (K=16 zero-padded to 32), concat into ab columns
#pragma unroll
    for (int nt = 0; nt < 4; ++nt) {
      v8f aa = {};
      aa = wmma_bf16(frag_A(&pb[wv][l16][0], 0, hf),
                     frag_B(&vt[wv][0][0], nt * 16 + l16, 32, 0, hf), aa);
#pragma unroll
      for (int r = 0; r < 8; ++r)
        ab[wv][r + 8 * hf][h * 64 + nt * 16 + l16] = f2bf(aa[r]);
    }
    LDS_FENCE();
  }

  // o = attn @ Wo + bo ; x = x + o   (A fragments hoisted from ab)
#pragma unroll
  for (int kk = 0; kk < 8; ++kk) afr[kk] = frag_A(&ab[wv][l16][0], kk * 32, hf);
#pragma unroll 4
  for (int nt = 0; nt < 16; ++nt) {
    v8f oa = {};
#pragma unroll
    for (int kk = 0; kk < 8; ++kk)
      oa = wmma_bf16(afr[kk], frag_B(woT, nt * 16 + l16, 256, kk * 32, hf), oa);
    float bias = bo[nt * 16 + l16];
#pragma unroll
    for (int r = 0; r < 8; ++r) {
      int m = r + 8 * hf, n = nt * 16 + l16;
      xs[wv][m][n] += oa[r] + bias;
    }
  }
  LDS_FENCE();

  // ---------------- MLP ----------------
  layernorm(g2, be2);
  LDS_FENCE();
#pragma unroll
  for (int kk = 0; kk < 8; ++kk) afr[kk] = frag_A(&hb[wv][l16][0], kk * 32, hf);

  v8f oacc[16];
#pragma unroll
  for (int t = 0; t < 16; ++t) oacc[t] = (v8f){};

  for (int jk = 0; jk < 32; ++jk) {  // FF in K=32 chunks, fused GEMM1->relu->GEMM2
#pragma unroll
    for (int jj = 0; jj < 2; ++jj) {
      int f = jk * 2 + jj;
      v8f aa = {};
#pragma unroll
      for (int kk = 0; kk < 8; ++kk)
        aa = wmma_bf16(afr[kk], frag_B(w1T, f * 16 + l16, 256, kk * 32, hf), aa);
      float bias = b1[f * 16 + l16];
#pragma unroll
      for (int r = 0; r < 8; ++r) {
        float v = fmaxf(aa[r] + bias, 0.f);
        actb[wv][r + 8 * hf][jj * 16 + l16] = f2bf(v);
      }
    }
    LDS_FENCE();
    v16bf act = frag_A(&actb[wv][l16][0], 0, hf);
#pragma unroll
    for (int t = 0; t < 16; ++t)
      oacc[t] = wmma_bf16(act, frag_B(w2T, t * 16 + l16, 1024, jk * 32, hf), oacc[t]);
    LDS_FENCE();
  }

  // out = x + mlp + b2
  float* ob = out + (size_t)b * (T_SEQ * C_DIM);
#pragma unroll 4
  for (int t = 0; t < 16; ++t) {
    float bias = b2[t * 16 + l16];
#pragma unroll
    for (int r = 0; r < 8; ++r) {
      int m = r + 8 * hf, n = t * 16 + l16;
      ob[m * 256 + n] = xs[wv][m][n] + oacc[t][r] + bias;
    }
  }
}

extern "C" void kernel_launch(void* const* d_in, const int* in_sizes, int n_in,
                              void* d_out, int out_size, void* d_ws, size_t ws_size,
                              hipStream_t stream) {
  (void)in_sizes; (void)n_in; (void)out_size; (void)ws_size;
  const float* x   = (const float*)d_in[0];
  const float* Wq  = (const float*)d_in[1];
  const float* Wk  = (const float*)d_in[2];
  const float* Wv  = (const float*)d_in[3];
  const float* Wo  = (const float*)d_in[4];
  const float* bo  = (const float*)d_in[5];
  const float* W1  = (const float*)d_in[6];
  const float* b1  = (const float*)d_in[7];
  const float* W2  = (const float*)d_in[8];
  const float* b2  = (const float*)d_in[9];
  const float* g1  = (const float*)d_in[10];
  const float* be1 = (const float*)d_in[11];
  const float* g2  = (const float*)d_in[12];
  const float* be2 = (const float*)d_in[13];
  float* out = (float*)d_out;

  // bf16 transposed weights in workspace (L2-resident, ~1.5MB)
  u16* ws  = (u16*)d_ws;
  u16* qT  = ws;                 // H*HS*C   = 65536
  u16* kT  = qT  + 65536;
  u16* vT  = kT  + 65536;
  u16* woT = vT  + 65536;        // 256*256  = 65536
  u16* w1T = woT + 65536;        // 1024*256 = 262144
  u16* w2T = w1T + 262144;       // 256*1024 = 262144

  const int TB = 256;
  conv_transpose_bf16<<<(65536 + TB - 1) / TB, TB, 0, stream>>>(Wq, qT, 256, 64, 65536);
  conv_transpose_bf16<<<(65536 + TB - 1) / TB, TB, 0, stream>>>(Wk, kT, 256, 64, 65536);
  conv_transpose_bf16<<<(65536 + TB - 1) / TB, TB, 0, stream>>>(Wv, vT, 256, 64, 65536);
  conv_transpose_bf16<<<(65536 + TB - 1) / TB, TB, 0, stream>>>(Wo, woT, 256, 256, 65536);
  conv_transpose_bf16<<<(262144 + TB - 1) / TB, TB, 0, stream>>>(W1, w1T, 256, 1024, 262144);
  conv_transpose_bf16<<<(262144 + TB - 1) / TB, TB, 0, stream>>>(W2, w2T, 1024, 256, 262144);

  block_fused<<<8192 / 4, 128, 0, stream>>>(x, qT, kT, vT, woT, w1T, w2T,
                                            bo, b1, b2, g1, be1, g2, be2, out);
}
